// DeepSeekMOE_22058952032744
// MI455X (gfx1250) — matrix operand order repaired and checked by reading, not compiled, not run on hardware
//
#include <hip/hip_runtime.h>
#include <hip/hip_bf16.h>
#include <math.h>

typedef __attribute__((ext_vector_type(4)))  __bf16 v4bf;
typedef __attribute__((ext_vector_type(8)))  __bf16 v8bf;
typedef __attribute__((ext_vector_type(16))) __bf16 v16bf;
typedef __attribute__((ext_vector_type(8)))  float  v8f;

union Frag { v16bf v16; v8bf v8[2]; };

#define T_TOK 4096
#define HDIM  1024
#define FDIM  2048
#define NEXP  8

#define BM 128
#define BN 128
#define BK 64
#define LSTR (BK + 8)    // 72 bf16 = 144B LDS row stride (pad for banks, /8 aligned)

#if defined(__AMDGCN__) && __has_builtin(__builtin_amdgcn_tensor_load_to_lds) && \
    __has_builtin(__builtin_amdgcn_s_wait_tensorcnt)
#define USE_TDM 1
#else
#define USE_TDM 0
#endif

#if USE_TDM
typedef __attribute__((ext_vector_type(4))) unsigned int u32x4;
typedef __attribute__((ext_vector_type(8))) int          i32x8;
typedef __attribute__((ext_vector_type(4))) int          i32x4;

// Issue one TDM descriptor: 2-D tensor [rows][K] bf16, tile BK x 128, into LDS
// with 8-element (4-dword) padding after every 64-element (32-dword) row ->
// LDS row stride = LSTR = 72 elements.
__device__ __forceinline__ void tdm_issue(const __bf16* g, const __bf16* ldsPtr,
                                          int K, int rows) {
  unsigned long long ga = (unsigned long long)(uintptr_t)g;
  unsigned ldsOff = (unsigned)(uintptr_t)ldsPtr;     // low 32 bits = LDS byte offset
  u32x4 g0;
  g0.x = 1u;                                          // count=1, user descriptor
  g0.y = ldsOff;                                      // lds_addr
  g0.z = (unsigned)ga;                                // global_addr[31:0]
  g0.w = (unsigned)((ga >> 32) & 0x01ffffffull) | 0x80000000u;  // addr[56:32] | type=2
  i32x8 g1;
  g1[0] = (int)((1u << 16)  |   // data_size = 2 bytes
                (1u << 20)  |   // pad_enable
                (4u << 22)  |   // pad_interval: 32 dwords
                (3u << 25));    // pad_amount:   4 dwords
  g1[1] = (int)(((unsigned)K & 0xffffu) << 16);                         // dim0[15:0]
  g1[2] = (int)((((unsigned)K >> 16) & 0xffffu) |
                (((unsigned)rows & 0xffffu) << 16));                    // dim0[31:16] | dim1[15:0]
  g1[3] = (int)((((unsigned)rows >> 16) & 0xffffu) | ((unsigned)BK << 16)); // dim1[31:16] | tile_dim0
  g1[4] = BM;                                        // tile_dim1=128, tile_dim2=0
  g1[5] = K;                                         // tensor_dim0_stride[31:0]
  g1[6] = 0;
  g1[7] = 0;
  i32x4 z4 = {0, 0, 0, 0};
  i32x8 z8 = {0, 0, 0, 0, 0, 0, 0, 0};
  // 6-arg toolchain variant (clang-23 / therock-10.0 headers)
  __builtin_amdgcn_tensor_load_to_lds(g0, g1, z4, z4, z8, 0);
}
#endif

// ---------------- utility kernels ----------------

__global__ __launch_bounds__(256) void zero_f32(float* __restrict__ p, int n) {
  int i4 = (blockIdx.x * 256 + threadIdx.x) * 4;
  if (i4 < n) *(float4*)(p + i4) = make_float4(0.f, 0.f, 0.f, 0.f);
}

__global__ __launch_bounds__(256) void cvt_bf16(const float* __restrict__ in,
                                                __bf16* __restrict__ out, int n) {
  int i4 = (blockIdx.x * 256 + threadIdx.x) * 4;
  if (i4 < n) {
    float4 v = *(const float4*)(in + i4);
    v4bf o;
    o[0] = (__bf16)v.x; o[1] = (__bf16)v.y; o[2] = (__bf16)v.z; o[3] = (__bf16)v.w;
    *(v4bf*)(out + i4) = o;
  }
}

// f32 [R][C] row-major  ->  bf16 [C][R]  (LDS-tiled 64x64 transpose + convert)
__global__ __launch_bounds__(256) void cvt_transpose(const float* __restrict__ in,
                                                     __bf16* __restrict__ out,
                                                     int R, int C) {
  __shared__ __bf16 tile[64][72];
  const int r0 = blockIdx.y * 64;
  const int c0 = blockIdx.x * 64;
  #pragma unroll
  for (int c = 0; c < 4; ++c) {
    int idx = c * 256 + threadIdx.x;     // 0..1023 float4 quads
    int r = idx >> 4;                    // 0..63
    int q = (idx & 15) * 4;              // 0..60
    float4 v = *(const float4*)(in + (size_t)(r0 + r) * C + c0 + q);
    tile[q + 0][r] = (__bf16)v.x;
    tile[q + 1][r] = (__bf16)v.y;
    tile[q + 2][r] = (__bf16)v.z;
    tile[q + 3][r] = (__bf16)v.w;
  }
  __syncthreads();
  #pragma unroll
  for (int c = 0; c < 2; ++c) {
    int idx = c * 256 + threadIdx.x;     // 0..511 v8bf chunks
    int cc = idx >> 3;                   // 0..63 output row
    int rr = (idx & 7) * 8;              // 0..56
    v8bf v = *(v8bf*)&tile[cc][rr];
    *(v8bf*)(out + (size_t)(c0 + cc) * R + r0 + rr) = v;
  }
}

// router: per-token logits + top-2 -> 0/1 mask [T, E]
__global__ __launch_bounds__(256) void router_kernel(const float* __restrict__ x,
                                                     const float* __restrict__ rw,
                                                     const float* __restrict__ rb,
                                                     float* __restrict__ mask,
                                                     int T, int H) {
  int t = blockIdx.x * 256 + threadIdx.x;
  if (t >= T) return;
  float logit[NEXP];
  #pragma unroll
  for (int e = 0; e < NEXP; ++e) logit[e] = rb[e];
  const float* xr = x + (size_t)t * H;
  for (int h = 0; h < H; ++h) {
    float xv = xr[h];
    const float* wr = rw + (size_t)h * NEXP;
    #pragma unroll
    for (int e = 0; e < NEXP; ++e) logit[e] += xv * wr[e];
  }
  int i1 = 0;
  #pragma unroll
  for (int e = 1; e < NEXP; ++e) if (logit[e] > logit[i1]) i1 = e;
  int i2 = (i1 == 0) ? 1 : 0;
  #pragma unroll
  for (int e = 0; e < NEXP; ++e)
    if (e != i1 && logit[e] > logit[i2]) i2 = e;
  #pragma unroll
  for (int e = 0; e < NEXP; ++e)
    mask[(size_t)t * NEXP + e] = (e == i1 || e == i2) ? 1.0f : 0.0f;
}

// ---------------- WMMA GEMM core ----------------

__device__ __forceinline__ void compute_tile(const __bf16 (*Asb)[LSTR],
                                             const __bf16 (*Bsb)[LSTR],
                                             int wm, int wn, int lr,
                                             int akb, int bkb, v8f (&acc)[2][4]) {
  #pragma unroll
  for (int ks = 0; ks < BK / 32; ++ks) {
    Frag a[2], b[4];
    #pragma unroll
    for (int i = 0; i < 2; ++i) {
      int row = wm + i * 16 + lr;
      a[i].v8[0] = *(const v8bf*)&Asb[row][ks * 32 + akb];
      a[i].v8[1] = *(const v8bf*)&Asb[row][ks * 32 + akb + 16];
    }
    #pragma unroll
    for (int j = 0; j < 4; ++j) {
      int col = wn + j * 16 + lr;
      b[j].v8[0] = *(const v8bf*)&Bsb[col][ks * 32 + bkb];
      b[j].v8[1] = *(const v8bf*)&Bsb[col][ks * 32 + bkb + 16];
    }
    #pragma unroll
    for (int i = 0; i < 2; ++i)
      #pragma unroll
      for (int j = 0; j < 4; ++j)
        acc[i][j] = __builtin_amdgcn_wmma_f32_16x16x32_bf16(
            false, a[i].v16, false, b[j].v16, (short)0, acc[i][j], false, false);
  }
}

__device__ __forceinline__ void load_tile_regs(const __bf16* Ab, const __bf16* Bb,
                                               int k0, int K, int tid,
                                               v8bf (&ra)[4], v8bf (&rb)[4]) {
  #pragma unroll
  for (int c = 0; c < 4; ++c) {
    int idx = c * 256 + tid;       // 0..1023 chunks of 8
    int row = idx >> 3;            // 0..127
    int col = (idx & 7) * 8;       // 0..56
    ra[c] = *(const v8bf*)(Ab + (size_t)row * K + k0 + col);
    rb[c] = *(const v8bf*)(Bb + (size_t)row * K + k0 + col);
  }
}

__device__ __forceinline__ void store_tile_lds(__bf16 (*Asb)[LSTR], __bf16 (*Bsb)[LSTR],
                                               int tid, const v8bf (&ra)[4],
                                               const v8bf (&rb)[4]) {
  #pragma unroll
  for (int c = 0; c < 4; ++c) {
    int idx = c * 256 + tid;
    int row = idx >> 3;
    int col = (idx & 7) * 8;
    *(v8bf*)&Asb[row][col] = ra[c];
    *(v8bf*)&Bsb[row][col] = rb[c];
  }
}

// C[M,N] = A[M,K] (bf16 row-major) * Bt[N,K]^T (bf16, K contiguous per row)
// MODE 0: out_bf16 = gelu_exact(C + bias)       MODE 1: out_f32 += scale[row]*(C + bias)
template <int MODE>
__global__ __launch_bounds__(256)
void moe_gemm(const __bf16* __restrict__ A, const __bf16* __restrict__ Bt,
              const float* __restrict__ bias,
              const float* __restrict__ scale, int scaleStride,
              __bf16* __restrict__ outBf, float* __restrict__ outF,
              int M, int N, int K) {
  __shared__ __bf16 As[2][BM][LSTR];
  __shared__ __bf16 Bs[2][BN][LSTR];

  const int tid  = threadIdx.x;
  const int bm   = blockIdx.y * BM;
  const int bn   = blockIdx.x * BN;
  const int lane = tid & 31;
  const int wave = tid >> 5;
  const int wm   = (wave & 3) * 32;   // 4 waves along M
  const int wn   = (wave >> 2) * 64;  // 2 waves along N
  const int lr   = lane & 15;
  const bool hi  = lane >= 16;
  const int akb  = hi ? 8 : 0;        // A frag: lanes0-15 K{0..7,16..23}; 16-31 K{8..15,24..31}
  const int bkb  = hi ? 8 : 0;        // B frag: lanes0-15 K{0..7,16..23}; 16-31 K{8..15,24..31}

  v8f acc[2][4];
  #pragma unroll
  for (int i = 0; i < 2; ++i)
    #pragma unroll
    for (int j = 0; j < 4; ++j)
      acc[i][j] = (v8f){0.f, 0.f, 0.f, 0.f, 0.f, 0.f, 0.f, 0.f};

  const __bf16* Ablk = A  + (size_t)bm * K;
  const __bf16* Bblk = Bt + (size_t)bn * K;
  const int nk = K / BK;

#if USE_TDM
  if (wave == 0) {
    tdm_issue(Ablk, &As[0][0][0], K, M);
    tdm_issue(Bblk, &Bs[0][0][0], K, N);
  }
  for (int i = 0; i < nk; ++i) {
    const int cur = i & 1;
    if (wave == 0) __builtin_amdgcn_s_wait_tensorcnt(0);   // tile i landed in LDS
    __syncthreads();                                       // visible to all waves
    if (wave == 0 && (i + 1) < nk) {                       // prefetch tile i+1 via TDM
      const int nxt = (i + 1) & 1;
      tdm_issue(Ablk + (size_t)(i + 1) * BK, &As[nxt][0][0], K, M);
      tdm_issue(Bblk + (size_t)(i + 1) * BK, &Bs[nxt][0][0], K, N);
    }
    compute_tile(As[cur], Bs[cur], wm, wn, lr, akb, bkb, acc);
  }
#else
  v8bf ra[4], rb[4];
  load_tile_regs(Ablk, Bblk, 0, K, tid, ra, rb);
  store_tile_lds(As[0], Bs[0], tid, ra, rb);
  for (int i = 0; i < nk; ++i) {
    __syncthreads();
    const int cur = i & 1;
    const bool pre = (i + 1) < nk;
    if (pre) load_tile_regs(Ablk, Bblk, (i + 1) * BK, K, tid, ra, rb);  // global loads first
    compute_tile(As[cur], Bs[cur], wm, wn, lr, akb, bkb, acc);          // hide latency
    if (pre) store_tile_lds(As[cur ^ 1], Bs[cur ^ 1], tid, ra, rb);
  }
#endif

  // epilogue: C layout VGPR r -> row = r + 8*(lane>=16), col = lane%16
  #pragma unroll
  for (int i = 0; i < 2; ++i) {
    #pragma unroll
    for (int j = 0; j < 4; ++j) {
      int col = bn + wn + j * 16 + lr;
      float bv = bias[col];
      #pragma unroll
      for (int r = 0; r < 8; ++r) {
        int row = bm + wm + i * 16 + r + (hi ? 8 : 0);
        float v = acc[i][j][r] + bv;
        if (MODE == 0) {
          float g = 0.5f * v * (1.0f + erff(v * 0.70710678118654752f));
          outBf[(size_t)row * N + col] = (__bf16)g;
        } else {
          float s = scale ? scale[(size_t)row * scaleStride] : 1.0f;
          if (s != 0.0f)
            outF[(size_t)row * N + col] += s * v;
        }
      }
    }
  }
}

// ---------------- launcher ----------------

extern "C" void kernel_launch(void* const* d_in, const int* in_sizes, int n_in,
                              void* d_out, int out_size, void* d_ws, size_t ws_size,
                              hipStream_t stream) {
  (void)in_sizes; (void)n_in; (void)out_size; (void)ws_size;
  const float* x   = (const float*)d_in[0];
  const float* sw1 = (const float*)d_in[1];
  const float* sb1 = (const float*)d_in[2];
  const float* sw2 = (const float*)d_in[3];
  const float* sb2 = (const float*)d_in[4];
  const float* ew1 = (const float*)d_in[5];
  const float* eb1 = (const float*)d_in[6];
  const float* ew2 = (const float*)d_in[7];
  const float* eb2 = (const float*)d_in[8];
  const float* rw  = (const float*)d_in[9];
  const float* rb  = (const float*)d_in[10];
  float* out = (float*)d_out;

  // workspace layout (~34 MB)
  char* ws = (char*)d_ws;
  __bf16* xb   = (__bf16*)ws;  ws += (size_t)T_TOK * HDIM * 2;
  __bf16* hid  = (__bf16*)ws;  ws += (size_t)T_TOK * FDIM * 2;
  __bf16* w1t  = (__bf16*)ws;  ws += (size_t)HDIM * FDIM * 2;  // [F][H]
  __bf16* w2t  = (__bf16*)ws;  ws += (size_t)FDIM * HDIM * 2;  // [H][F]
  float*  mask = (float*)ws;

  const int nOut = T_TOK * HDIM;
  zero_f32<<<(nOut / 4 + 255) / 256, 256, 0, stream>>>(out, nOut);
  cvt_bf16<<<((T_TOK * HDIM) / 4 + 255) / 256, 256, 0, stream>>>(x, xb, T_TOK * HDIM);
  router_kernel<<<(T_TOK + 255) / 256, 256, 0, stream>>>(x, rw, rb, mask, T_TOK, HDIM);

  dim3 g1(FDIM / BN, T_TOK / BM);          // 16 x 32
  dim3 g2(HDIM / BN, T_TOK / BM);          //  8 x 32
  dim3 t1(FDIM / 64, HDIM / 64);           // transpose w1 [H][F] -> [F][H]
  dim3 t2(HDIM / 64, FDIM / 64);           // transpose w2 [F][H] -> [H][F]

  for (int i = 0; i < 10; ++i) {
    const float *w1, *b1, *w2, *b2, *scale;
    if (i < 2) {
      w1 = sw1 + (size_t)i * HDIM * FDIM;  b1 = sb1 + (size_t)i * FDIM;
      w2 = sw2 + (size_t)i * FDIM * HDIM;  b2 = sb2 + (size_t)i * HDIM;
      scale = nullptr;
    } else {
      int e = i - 2;
      w1 = ew1 + (size_t)e * HDIM * FDIM;  b1 = eb1 + (size_t)e * FDIM;
      w2 = ew2 + (size_t)e * FDIM * HDIM;  b2 = eb2 + (size_t)e * HDIM;
      scale = mask + e;
    }
    cvt_transpose<<<t1, 256, 0, stream>>>(w1, w1t, HDIM, FDIM);
    cvt_transpose<<<t2, 256, 0, stream>>>(w2, w2t, FDIM, HDIM);
    // hid = gelu(x @ w1 + b1)
    moe_gemm<0><<<g1, 256, 0, stream>>>(xb, w1t, b1, nullptr, 0, hid, nullptr,
                                        T_TOK, FDIM, HDIM);
    // out += scale * (hid @ w2 + b2)
    moe_gemm<1><<<g2, 256, 0, stream>>>(hid, w2t, b2, scale, NEXP, nullptr, out,
                                        T_TOK, HDIM, FDIM);
  }
}
